// AT_BLOCK_26414048871161
// MI455X (gfx1250) — matrix operand order repaired and checked by reading, not compiled, not run on hardware
//
#include <hip/hip_runtime.h>

// ---------------- problem constants (from reference setup_inputs) ----------------
#define B_   32      // batch
#define HD   512     // h_dim (reduction K of scores matmul)
#define TD   256     // t_dim (reduction K of projection)
#define MD   128     // M (words; softmax axis)
#define ND   4096    // N = h*w

// ---------------- CDNA5 WMMA types ----------------
typedef __attribute__((ext_vector_type(16))) __bf16 v16bf;
typedef __attribute__((ext_vector_type(8)))  float  v8f;
typedef __attribute__((ext_vector_type(4)))  int    v4i;

union BF16x16 { v16bf v; __bf16 h[16]; v4i q[2]; };

// Native f32 -> bf16 convert (v_cvt_*bf16* on gfx1250) instead of manual RNE bit math.
__device__ __forceinline__ __bf16 f2bf(float f) { return (__bf16)f; }

__device__ __forceinline__ v8f wmma_bf16(const BF16x16& a, const BF16x16& b, v8f c) {
  // D(f32 16x16) = A(bf16 16x32) x B(bf16 32x16) + C
  return __builtin_amdgcn_wmma_f32_16x16x32_bf16(false, a.v, false, b.v,
                                                 (short)0, c, false, false);
}

// ---------------- async global->LDS (CDNA5 ASYNCcnt path) ----------------
#define USE_ASYNC_G2L 1
#if USE_ASYNC_G2L
__device__ __forceinline__ void g2l_b128(unsigned lds_off, unsigned long long gaddr) {
  asm volatile("global_load_async_to_lds_b128 %0, %1, off"
               :: "v"(lds_off), "v"(gaddr) : "memory");
}
__device__ __forceinline__ void wait_async0() {
  asm volatile("s_wait_asynccnt 0x0" ::: "memory");
}
#endif

// ---------------- LDS layout for fused kernel (padded vs 64x4B banks) ----------------
#define WP_STRIDE  136                              // 128 + 8 bf16 pad -> +16B/row
#define WPT_STRIDE 520                              // 512 + 8 bf16 pad -> +16B/row
#define PB_STRIDE  136
#define WP_OFF     0
#define WPT_OFF    (HD * WP_STRIDE * 2)             // 139264
#define PB_OFF     (WPT_OFF + MD * WPT_STRIDE * 2)  // 272384
#define SMEM_BYTES (PB_OFF + 8 * 16 * PB_STRIDE * 2)// 307200 (<320KB WGP LDS)

// =================================================================================
// Kernel 1: Wp[b] = head_w @ W[b]   ([512,256]x[256,128]) -> bf16 Wp and WpT
// One wave per 16x16 output tile, 8 waves / block.
// =================================================================================
__global__ void proj_1x1(const float* __restrict__ W, const float* __restrict__ head_w,
                         __bf16* __restrict__ Wp, __bf16* __restrict__ WpT)
{
  const int wid  = (blockIdx.x * blockDim.x + threadIdx.x) >> 5;
  const int lane = threadIdx.x & 31;
  const int b    = wid >> 8;          // 256 tiles per batch
  const int t    = wid & 255;
  const int h0   = (t >> 3) << 4;     // 32 tiles along h_dim
  const int m0   = (t & 7)  << 4;     // 8 tiles along M
  const int row  = lane & 15;
  const int hf   = lane >> 4;
  const int kbA  = hf * 16;           // A-matrix per-lane K base (contiguous K)
  const int kbB  = hf * 16;           // B-matrix per-lane K base

  const float* Wb = W + (size_t)b * TD * MD;
  v8f c = {};
  for (int kt = 0; kt < TD / 32; ++kt) {
    BF16x16 a, bb;
    // A: head_w rows (contiguous K) -> base + immediate offsets
    const float* ap = head_w + (h0 + row) * TD + kt * 32 + kbA;
    // B: W[b] column m0+row, K strided by MD -> base + immediate offsets
    const float* bp = Wb + (size_t)(kt * 32 + kbB) * MD + m0 + row;
#pragma unroll
    for (int j = 0; j < 8; ++j) {
      a.h[j]     = f2bf(ap[j]);
      a.h[j + 8] = f2bf(ap[8 + j]);
    }
#pragma unroll
    for (int j = 0; j < 16; ++j)
      bb.h[j] = f2bf(bp[j * MD]);
    c = wmma_bf16(a, bb, c);
  }
#pragma unroll
  for (int r = 0; r < 8; ++r) {
    const int h = h0 + r + hf * 8;    // D-layout: rows split across lane halves
    const int m = m0 + row;
    const __bf16 v = f2bf(c[r]);
    Wp [((size_t)b * HD + h) * MD + m] = v;
    WpT[((size_t)b * MD + m) * HD + h] = v;
  }
}

// =================================================================================
// Kernel 2: fused scores -> softmax -> context, one block per (batch, 128-row n-tile)
// =================================================================================
__global__ void __launch_bounds__(256, 1)
fused_attn(const float* __restrict__ H, const __bf16* __restrict__ WpG,
           const __bf16* __restrict__ WpTG, float* __restrict__ out)
{
  extern __shared__ char smem[];
  const int tid  = threadIdx.x;
  const int b    = blockIdx.x >> 5;
  const int nt   = blockIdx.x & 31;
  const int wave = tid >> 5;
  const int lane = tid & 31;
  const int row  = lane & 15;
  const int hf   = lane >> 4;
  const int kbA  = hf * 16;
  const int kbB  = hf * 16;
  const int n0   = nt * 128 + wave * 16;   // this wave's 16 rows of N

  // ---- stage Wp[512x128] and WpT[128x512] (bf16) into padded LDS ----
  const __bf16* wp  = WpG  + (size_t)b * HD * MD;
  const __bf16* wpt = WpTG + (size_t)b * MD * HD;
  for (int cidx = tid; cidx < (HD * MD) / 8; cidx += 256) {
    const int e = cidx * 8;                      // 8 bf16 = 16B per chunk
    const int r1 = e >> 7, c1 = e & 127;         // Wp  row/col
    const int r2 = e >> 9, c2 = e & 511;         // WpT row/col
#if USE_ASYNC_G2L
    g2l_b128((unsigned)(size_t)(smem + WP_OFF  + (r1 * WP_STRIDE  + c1) * 2),
             (unsigned long long)(wp + e));
    g2l_b128((unsigned)(size_t)(smem + WPT_OFF + (r2 * WPT_STRIDE + c2) * 2),
             (unsigned long long)(wpt + e));
#else
    *(v4i*)(smem + WP_OFF  + (r1 * WP_STRIDE  + c1) * 2) = *(const v4i*)(wp  + e);
    *(v4i*)(smem + WPT_OFF + (r2 * WPT_STRIDE + c2) * 2) = *(const v4i*)(wpt + e);
#endif
  }
#if USE_ASYNC_G2L
  wait_async0();
#endif
  __syncthreads();

  // ---- phase 1: scores[16 x 128] = H^T tile @ Wp  (K = 512, bf16 WMMA) ----
  const float* Hb = H + (size_t)b * HD * ND;
  v8f acc[8];
#pragma unroll
  for (int mt = 0; mt < 8; ++mt) acc[mt] = (v8f){};

  for (int kt = 0; kt < HD / 32; ++kt) {
    BF16x16 a;
    const int hbase = kt * 32;
    // single base pointer per K-step; all 16 loads use immediate offsets (< 8MB)
    const float* hp = Hb + ((size_t)hbase + kbA) * ND + n0 + row;
#pragma unroll
    for (int j = 0; j < 8; ++j) {
      a.h[j]     = f2bf(hp[(size_t)j * ND]);
      a.h[j + 8] = f2bf(hp[(size_t)(8 + j) * ND]);
    }
#pragma unroll
    for (int mt = 0; mt < 8; ++mt) {          // B fragments from LDS WpT[m][h]
      BF16x16 bf;
      const char* base = smem + WPT_OFF + ((mt * 16 + row) * WPT_STRIDE + hbase + kbB) * 2;
      bf.q[0] = *(const v4i*)(base);
      bf.q[1] = *(const v4i*)(base + 16);
      acc[mt] = wmma_bf16(a, bf, acc[mt]);
    }
  }

  // ---- softmax over M=128 (f32, in registers; rows live per lane-half) ----
#pragma unroll
  for (int r = 0; r < 8; ++r) {
    float mx = acc[0][r];
#pragma unroll
    for (int mt = 1; mt < 8; ++mt) mx = fmaxf(mx, acc[mt][r]);
    for (int o = 8; o; o >>= 1) mx = fmaxf(mx, __shfl_xor(mx, o, 32));
    float s = 0.f;
#pragma unroll
    for (int mt = 0; mt < 8; ++mt) { float e = __expf(acc[mt][r] - mx); acc[mt][r] = e; s += e; }
    for (int o = 8; o; o >>= 1) s += __shfl_xor(s, o, 32);
    const float inv = 1.0f / s;
#pragma unroll
    for (int mt = 0; mt < 8; ++mt) acc[mt][r] *= inv;
  }

  // ---- transpose probs (D layout -> B layout) through per-wave LDS scratch ----
  char* pbase = smem + PB_OFF + wave * 16 * PB_STRIDE * 2;   // [16 n][128 m] bf16
#pragma unroll
  for (int mt = 0; mt < 8; ++mt)
#pragma unroll
    for (int r = 0; r < 8; ++r) {
      const int nl = r + hf * 8;
      const int m  = mt * 16 + row;
      *(__bf16*)(pbase + (nl * PB_STRIDE + m) * 2) = f2bf(acc[mt][r]);
    }

  BF16x16 pb[4];                               // B fragments of probs^T (K=m), kept live
#pragma unroll
  for (int mt = 0; mt < 4; ++mt) {
    const char* base = pbase + (row * PB_STRIDE + mt * 32 + kbB) * 2;
    pb[mt].q[0] = *(const v4i*)(base);
    pb[mt].q[1] = *(const v4i*)(base + 16);
  }

  // ---- phase 2: out^T tiles = Wp @ probs^T  (K = 128) -> direct [h][n] stores ----
  float* outb = out + (size_t)b * HD * ND;
  for (int ht = 0; ht < HD / 16; ++ht) {
    v8f c = {};
#pragma unroll
    for (int mt = 0; mt < 4; ++mt) {
      BF16x16 a;                               // A fragment = Wp rows from LDS
      const char* base = smem + WP_OFF + ((ht * 16 + row) * WP_STRIDE + mt * 32 + kbA) * 2;
      a.q[0] = *(const v4i*)(base);
      a.q[1] = *(const v4i*)(base + 16);       // K = kbA+8..kbA+15 of the fragment
      c = wmma_bf16(a, pb[mt], c);
    }
    // D rows = h, cols = n: per-r stores are 64B-contiguous across each lane half
    float* op = outb + (size_t)(ht * 16 + hf * 8) * ND + n0 + row;
#pragma unroll
    for (int r = 0; r < 8; ++r)
      op[(size_t)r * ND] = c[r];
  }
}

// =================================================================================
extern "C" void kernel_launch(void* const* d_in, const int* in_sizes, int n_in,
                              void* d_out, int out_size, void* d_ws, size_t ws_size,
                              hipStream_t stream) {
  const float* H  = (const float*)d_in[0];   // [32,512,64,64]
  const float* W  = (const float*)d_in[1];   // [32,256,128]
  const float* hw = (const float*)d_in[2];   // [512,256]
  float* out = (float*)d_out;                // [32,512,4096]

  __bf16* Wp  = (__bf16*)d_ws;                       // bf16 [32][512][128]
  __bf16* WpT = Wp + (size_t)B_ * HD * MD;           // bf16 [32][128][512]

  // projection: 32*256 tiles, 8 waves/block -> 1024 blocks
  proj_1x1<<<dim3((B_ * 256) / 8), dim3(256), 0, stream>>>(W, hw, Wp, WpT);

  (void)hipFuncSetAttribute((const void*)fused_attn,
                            hipFuncAttributeMaxDynamicSharedMemorySize, SMEM_BYTES);
  // one block per (batch, 128-row n-tile): 32*32 = 1024 blocks, 8 waves each
  fused_attn<<<dim3(B_ * 32), dim3(256), SMEM_BYTES, stream>>>(H, Wp, WpT, out);

  (void)in_sizes; (void)n_in; (void)out_size; (void)ws_size;
}
